// MultiheadSelfAttention_50732153700904
// MI455X (gfx1250) — matrix operand order repaired and checked by reading, not compiled, not run on hardware
//
#include <hip/hip_runtime.h>

typedef __attribute__((ext_vector_type(16))) _Float16 v16h;
typedef __attribute__((ext_vector_type(8)))  float    v8f;
typedef __attribute__((ext_vector_type(4)))  unsigned v4u;
typedef __attribute__((ext_vector_type(8)))  unsigned v8u;
typedef _Float16 HT;

union FragU {
  v16h v;
  HT h[16];
  unsigned u[8];
};

// A-matrix 16x32 f16 fragment (CDNA5 layout): lane l (0-15) holds row M=l.
// VGPR v<4 : K = 2v+{0,1} (+8 for hi lanes); v>=4 : K = 16+2(v-4)+{0,1} (+8 for hi).
// p0 = tile base + row_base*stride + k_base ; stride in halfs (even).
__device__ __forceinline__ v16h load_fragA(const HT* p0, int stride, int l, int hi) {
  FragU f;
  const HT* p = p0 + l * stride + hi * 8;
#pragma unroll
  for (int v = 0; v < 4; ++v) {
    f.u[v]     = *(const unsigned*)(p + 2 * v);
    f.u[v + 4] = *(const unsigned*)(p + 16 + 2 * v);
  }
  return f.v;
}

// B-matrix 32x16 f16 fragment: lane l holds column N=l; VGPR v<4 : K = 2v+{0,1} (+8 for hi); v>=4 : K = 16+2(v-4)+{0,1} (+8 for hi).
// Source stored K-contiguous per column: p0 = base + col_base*stride + k_base.
__device__ __forceinline__ v16h load_fragB(const HT* p0, int stride, int l, int hi) {
  FragU f;
  const HT* p = p0 + l * stride + hi * 8;
#pragma unroll
  for (int v = 0; v < 4; ++v) {
    f.u[v]     = *(const unsigned*)(p + 2 * v);
    f.u[v + 4] = *(const unsigned*)(p + 16 + 2 * v);
  }
  return f.v;
}

__device__ __forceinline__ v8f wmma16(v16h a, v16h b, v8f c) {
  return __builtin_amdgcn_wmma_f32_16x16x32_f16(false, a, false, b, (short)0, c, false, false);
}

// One-instruction lane exchange: ds_swizzle group-of-32 XOR pattern
// (offset[4:0]=and_mask=0x1F, offset[14:10]=xor_mask). EXEC is all-ones here.
template <int XOR>
__device__ __forceinline__ float swz_xor(float x) {
  return __int_as_float(__builtin_amdgcn_ds_swizzle(__float_as_int(x), (XOR << 10) | 0x1F));
}
__device__ __forceinline__ float red_max16(float x) {
  x = fmaxf(x, swz_xor<1>(x));
  x = fmaxf(x, swz_xor<2>(x));
  x = fmaxf(x, swz_xor<4>(x));
  x = fmaxf(x, swz_xor<8>(x));
  return x;
}
__device__ __forceinline__ float red_sum16(float x) {
  x += swz_xor<1>(x);
  x += swz_xor<2>(x);
  x += swz_xor<4>(x);
  x += swz_xor<8>(x);
  return x;
}

// Issue a 2-D TDM load: 64x64 f16 tile, row stride `row_stride_elems`, into LDS
// at byte offset lds_off (linear, rows of 64 halfs). D# per CDNA5 ISA 8.3/8.4,
// VADDR2/VADDR3 omitted (<=2D tensor). Wave-level op; EXEC ignored; TENSORcnt.
__device__ __forceinline__ void tdm_load_64x64(unsigned lds_off, const HT* gptr,
                                               unsigned row_stride_elems) {
  const unsigned long long ga = (unsigned long long)(uintptr_t)gptr;
  v4u g0;
  g0[0] = 1u;                                   // count=1, user descriptor
  g0[1] = lds_off;                              // lds_addr (bytes)
  g0[2] = (unsigned)ga;                         // global_addr[31:0]
  g0[3] = (unsigned)(ga >> 32) | 0x80000000u;   // global_addr[56:32] | type=2
  v8u g1;
  g1[0] = 0x10000u;        // wg_mask=0, data_size=1 (2 bytes), no flags
  g1[1] = 64u << 16;       // tensor_dim0 = 64   (bits 79:48, low half)
  g1[2] = 64u << 16;       // tensor_dim0 hi = 0 | tensor_dim1 = 64 (lo16)
  g1[3] = 64u << 16;       // tensor_dim1 hi = 0 | tile_dim0 = 64
  g1[4] = 64u;             // tile_dim1 = 64, tile_dim2 = 0
  g1[5] = row_stride_elems;// tensor_dim0_stride[31:0] (data_size units)
  g1[6] = 0u;              // stride hi | tensor_dim1_stride lo (unused, 2D)
  g1[7] = 0u;
  asm volatile("tensor_load_to_lds %0, %1" :: "s"(g0), "s"(g1) : "memory");
}

// ---------------------------------------------------------------------------
// Tiled GEMM: C[M,N] = A[M,K] * B[K,N] + bias[N]
// MODE 0: A = fp32 x; epilogue scatters f16 Q(scaled)/K -> [B,H,T,D], V -> [B,H,D,T]
// MODE 1: A = f16 attention output; epilogue writes fp32 C
// Block: 256 threads (8 waves), tile 128x128, K-step 32; wave = 32x64 via 2x4 WMMAs.
// Register double buffer: next K-slice global loads overlap WMMA work.
// ---------------------------------------------------------------------------
template <int MODE>
__global__ __launch_bounds__(256) void gemm_wmma(
    const float* __restrict__ Af, const HT* __restrict__ Ah,
    const float* __restrict__ Bg, const float* __restrict__ bias,
    HT* __restrict__ qo, HT* __restrict__ ko, HT* __restrict__ vo,
    float* __restrict__ fo, int M, int N, int K) {
  __shared__ alignas(16) HT sA[128 * 48];  // [m][k] stride 48 halfs
  __shared__ alignas(16) HT sB[128 * 48];  // [n][k] stride 48 halfs (B transposed)

  const int tid = threadIdx.x;
  const int lane = tid & 31, w = tid >> 5;
  const int l = lane & 15, hi = lane >> 4;
  const int wm = (w & 3) * 32, wn = (w >> 2) * 64;
  const int M0 = blockIdx.y * 128, N0 = blockIdx.x * 128;
  const int rowA = tid >> 1, koffA = (tid & 1) * 16;

  v8f zero = {};
  v8f acc[2][4];
#pragma unroll
  for (int mt = 0; mt < 2; ++mt)
#pragma unroll
    for (int nt = 0; nt < 4; ++nt) acc[mt][nt] = zero;

  float aStageF[MODE == 0 ? 16 : 1];
  HT    aStageH[MODE == 1 ? 16 : 1];
  float bStage[16];

  auto stage_load = [&](int kb) {
    if constexpr (MODE == 0) {
      const float* src = Af + (size_t)(M0 + rowA) * K + kb + koffA;
#pragma unroll
      for (int i = 0; i < 16; ++i) aStageF[i] = src[i];
    } else {
      const HT* src = Ah + (size_t)(M0 + rowA) * K + kb + koffA;
#pragma unroll
      for (int i = 0; i < 16; ++i) aStageH[i] = src[i];
    }
#pragma unroll
    for (int i = 0; i < 16; ++i) {
      const int e = (i << 8) + tid;
      bStage[i] = Bg[(size_t)(kb + (e >> 7)) * N + N0 + (e & 127)];
    }
  };

  stage_load(0);
  for (int kb = 0; kb < K; kb += 32) {
    {  // commit staged slice to LDS (f32 -> f16)
      HT* dst = sA + rowA * 48 + koffA;
#pragma unroll
      for (int i = 0; i < 16; ++i)
        dst[i] = (MODE == 0) ? (HT)aStageF[i] : aStageH[i];
#pragma unroll
      for (int i = 0; i < 16; ++i) {
        const int e = (i << 8) + tid;
        sB[(e & 127) * 48 + (e >> 7)] = (HT)bStage[i];
      }
    }
    __syncthreads();
    if (kb + 32 < K) stage_load(kb + 32);  // global fetch overlaps WMMAs below

    v16h af[2], bf[4];
#pragma unroll
    for (int mt = 0; mt < 2; ++mt) af[mt] = load_fragA(sA + (wm + mt * 16) * 48, 48, l, hi);
#pragma unroll
    for (int nt = 0; nt < 4; ++nt) bf[nt] = load_fragB(sB + (wn + nt * 16) * 48, 48, l, hi);
#pragma unroll
    for (int mt = 0; mt < 2; ++mt)
#pragma unroll
      for (int nt = 0; nt < 4; ++nt) acc[mt][nt] = wmma16(af[mt], bf[nt], acc[mt][nt]);
    __syncthreads();
  }

  // Epilogue. C/D layout: VGPR r holds M = r (lanes 0-15) / r+8 (lanes 16-31), N = l.
#pragma unroll
  for (int mt = 0; mt < 2; ++mt)
#pragma unroll
    for (int nt = 0; nt < 4; ++nt)
#pragma unroll
      for (int r = 0; r < 8; ++r) {
        const int m = M0 + wm + mt * 16 + r + 8 * hi;
        const int n = N0 + wn + nt * 16 + l;
        const float val = acc[mt][nt][r] + bias[n];
        if constexpr (MODE == 0) {
          const int sect = n >> 11, within = n & 2047;
          const int hh = within >> 6, dd = within & 63;
          const int b = m >> 11, t = m & 2047;
          if (sect == 0)       // fold SCALE = 8/D into Q; layout [B,H,T,D]
            qo[(((size_t)(b * 32 + hh)) * 2048 + t) * 64 + dd] = (HT)(val * 0.125f);
          else if (sect == 1)  // K layout [B,H,T,D]
            ko[(((size_t)(b * 32 + hh)) * 2048 + t) * 64 + dd] = (HT)val;
          else                 // V layout [B,H,D,T]: pre-transposed for the P*V TDM tile
            vo[(((size_t)(b * 32 + hh)) * 64 + dd) * 2048 + t] = (HT)val;
        } else {
          fo[(size_t)m * N + n] = val;
        }
      }
}

// ---------------------------------------------------------------------------
// Flash attention, causal + ALiBi. One block per (b*H+h, 128-query tile).
// 8 waves; wave w owns query rows [q0+16w, q0+16w+16). 64-key blocks up to the
// causal frontier. K/V tiles staged by the Tensor Data Mover into double-
// buffered LDS: wave 0 waits tile kb, one barrier publishes, then issues tile
// kb+1 so the DMA runs under the WMMA + softmax work.
// ---------------------------------------------------------------------------
__global__ __launch_bounds__(256) void attn_wmma(
    const HT* __restrict__ Qg, const HT* __restrict__ Kg,
    const HT* __restrict__ Vg, HT* __restrict__ Og) {
  __shared__ alignas(16) HT sK[2 * 64 * 64];  // [buf][key][d]  (B-frag src, Q*K^T)
  __shared__ alignas(16) HT sV[2 * 64 * 64];  // [buf][d][key]  (B-frag src, P*V)
  __shared__ alignas(16) HT sP[8 * 16 * 72];  // per-wave P strip [m][key]

  const int tid = threadIdx.x;
  const int lane = tid & 31, w = tid >> 5;
  const int l = lane & 15, hi = lane >> 4;
  const int bh = blockIdx.y;
  const int h = bh & 31;
  const int q0 = blockIdx.x * 128;
  const int qrow0 = q0 + w * 16;
  const float slope = exp2f(-0.25f * (float)(h + 1));  // 2^(-8(h+1)/32)
  const size_t base = (size_t)bh * 2048 * 64;
  const HT* Qb = Qg + base;   // [T,64]
  const HT* Kb = Kg + base;   // [T,64]
  const HT* Vb = Vg + base;   // [64,T]  (d-major)
  HT* P = sP + w * (16 * 72);
  const unsigned sK_off = (unsigned)(uintptr_t)(void*)sK;
  const unsigned sV_off = (unsigned)(uintptr_t)(void*)sV;

  // Q fragments straight from global ([T,64] row-major matches A-frag gather)
  v16h qf[2];
  qf[0] = load_fragA(Qb + (size_t)qrow0 * 64, 64, l, hi);
  qf[1] = load_fragA(Qb + (size_t)qrow0 * 64 + 32, 64, l, hi);

  v8f zero = {};
  v8f o[4] = {zero, zero, zero, zero};
  float rmax[8], rsum[8], g[8], gn[4];
#pragma unroll
  for (int r = 0; r < 8; ++r) {
    rmax[r] = -1e30f;
    rsum[r] = 0.0f;
    // ALiBi row term: slope * (l - i); full bias = g[r] + gn[nt], sign == sign(j-i)
    g[r] = slope * (float)(l - (qrow0 + r + 8 * hi));
  }
#pragma unroll
  for (int nt = 0; nt < 4; ++nt) gn[nt] = slope * (float)(nt * 16);

  const int nkb = (q0 >> 6) + 2;  // causal: keys 0 .. q0+127 only
  if (w == 0) {                   // prologue: DMA tile 0 into buffer 0
    tdm_load_64x64(sK_off, Kb, 64u);
    tdm_load_64x64(sV_off, Vb, 2048u);
  }
  for (int kb = 0; kb < nkb; ++kb) {
    const int buf = kb & 1;
    const HT* tK = sK + buf * 4096;
    const HT* tV = sV + buf * 4096;
    if (w == 0) __builtin_amdgcn_s_wait_tensorcnt(0);  // tile kb landed
    __syncthreads();  // publish tile kb; all waves done with buf^1 from kb-1
    if (w == 0 && kb + 1 < nkb) {  // prefetch tile kb+1 under the math below
      const unsigned boff = (buf ^ 1) * 8192u;
      tdm_load_64x64(sK_off + boff, Kb + (size_t)(kb + 1) * 64 * 64, 64u);
      tdm_load_64x64(sV_off + boff, Vb + (kb + 1) * 64, 2048u);
    }

    // S = (Q*scale) K^T : 4 key sub-tiles, K-dim 64 = 2 WMMA steps each
    v8f s[4] = {zero, zero, zero, zero};
#pragma unroll
    for (int nt = 0; nt < 4; ++nt) {
      s[nt] = wmma16(qf[0], load_fragB(tK + nt * 16 * 64, 64, l, hi), s[nt]);
      s[nt] = wmma16(qf[1], load_fragB(tK + nt * 16 * 64 + 32, 64, l, hi), s[nt]);
    }

    // ALiBi + causal mask (reference: add bias, then where(j>i, -1e4)); bias>0 <=> j>i
    float bm[8];
#pragma unroll
    for (int r = 0; r < 8; ++r) bm[r] = -1e30f;
#pragma unroll
    for (int nt = 0; nt < 4; ++nt)
#pragma unroll
      for (int r = 0; r < 8; ++r) {
        const float bias = g[r] + gn[nt];
        float v = s[nt][r] + bias;
        if (bias > 0.0f) v = -10000.0f;
        s[nt][r] = v;
        bm[r] = fmaxf(bm[r], v);
      }
#pragma unroll
    for (int nt = 0; nt < 4; ++nt) gn[nt] += slope * 64.0f;  // advance key block
    // per-row reductions across the 16 lanes of a half-wave: single ds_swizzle per step
#pragma unroll
    for (int r = 0; r < 8; ++r) bm[r] = red_max16(bm[r]);

    float alpha[8], bs[8];
#pragma unroll
    for (int r = 0; r < 8; ++r) {
      const float mn = fmaxf(rmax[r], bm[r]);
      alpha[r] = __expf(rmax[r] - mn);
      rmax[r] = mn;
      bs[r] = 0.0f;
    }
#pragma unroll
    for (int nt = 0; nt < 4; ++nt)
#pragma unroll
      for (int r = 0; r < 8; ++r) {
        const float p = __expf(s[nt][r] - rmax[r]);
        s[nt][r] = p;
        bs[r] += p;
      }
#pragma unroll
    for (int r = 0; r < 8; ++r) {
      bs[r] = red_sum16(bs[r]);
      rsum[r] = rsum[r] * alpha[r] + bs[r];
    }
#pragma unroll
    for (int nt = 0; nt < 4; ++nt)
#pragma unroll
      for (int r = 0; r < 8; ++r) o[nt][r] *= alpha[r];

    // C-layout -> A-layout for P via this wave's private LDS strip
#pragma unroll
    for (int nt = 0; nt < 4; ++nt)
#pragma unroll
      for (int r = 0; r < 8; ++r)
        P[(r + 8 * hi) * 72 + nt * 16 + l] = (HT)s[nt][r];
    __builtin_amdgcn_wave_barrier();                // keep compiler from hoisting loads
    asm volatile("s_wait_dscnt 0" ::: "memory");    // DS in-order per wave; data visible

    v16h pf[2];
    pf[0] = load_fragA(P, 72, l, hi);
    pf[1] = load_fragA(P + 32, 72, l, hi);
#pragma unroll
    for (int nt = 0; nt < 4; ++nt) {
      o[nt] = wmma16(pf[0], load_fragB(tV + nt * 16 * 64, 64, l, hi), o[nt]);
      o[nt] = wmma16(pf[1], load_fragB(tV + nt * 16 * 64 + 32, 64, l, hi), o[nt]);
    }
  }

  // normalize and write O as f16 [B*T, E] row (col = h*64 + d) for the out-GEMM
  const int b = bh >> 5;
#pragma unroll
  for (int nt = 0; nt < 4; ++nt)
#pragma unroll
    for (int r = 0; r < 8; ++r) {
      const int t = qrow0 + r + 8 * hi;
      const int col = h * 64 + nt * 16 + l;
      Og[((size_t)(b * 2048 + t)) * 2048 + col] = (HT)(o[nt][r] / rsum[r]);
    }
}

// ---------------------------------------------------------------------------
extern "C" void kernel_launch(void* const* d_in, const int* in_sizes, int n_in,
                              void* d_out, int out_size, void* d_ws, size_t ws_size,
                              hipStream_t stream) {
  (void)in_sizes; (void)n_in; (void)out_size; (void)ws_size;
  const float* x    = (const float*)d_in[0];
  const float* Wqkv = (const float*)d_in[1];
  const float* bqkv = (const float*)d_in[2];
  const float* Wout = (const float*)d_in[3];
  const float* bout = (const float*)d_in[4];
  // d_in[5] attention_mask: known causal, recomputed on the fly
  float* out = (float*)d_out;

  const size_t elems = (size_t)2 * 2048 * 2048;  // B*T*E
  HT* Qh = (HT*)d_ws;
  HT* Kh = Qh + elems;
  HT* Vh = Kh + elems;  // stored [B,H,D,T]
  HT* Oh = Vh + elems;  // total ws use: 4 * 16 MiB = 64 MiB

  // 1) QKV projection: [4096,2048] x [2048,6144]
  gemm_wmma<0><<<dim3(48, 32), 256, 0, stream>>>(
      x, nullptr, Wqkv, bqkv, Qh, Kh, Vh, nullptr, 4096, 6144, 2048);
  // 2) causal flash attention with ALiBi: grid (T/128, B*H)
  attn_wmma<<<dim3(16, 64), 256, 0, stream>>>(Qh, Kh, Vh, Oh);
  // 3) output projection: [4096,2048] x [2048,2048] -> fp32
  gemm_wmma<1><<<dim3(16, 32), 256, 0, stream>>>(
      nullptr, Oh, Wout, bout, nullptr, nullptr, nullptr, out, 4096, 2048, 2048);
}
